// HashingMemory_4526895530560
// MI455X (gfx1250) — compile-verified, matrix operand order checked by reading, not run on hardware
//
#include <hip/hip_runtime.h>
#include <hip/hip_bf16.h>

#define HEADS  4
#define KDIM   256
#define HALF   128
#define NKEYS  512
#define KNN    32
#define VDIM   1024
#define IN_DIM 1024
#define F_DIM  (HEADS * KDIM)   // 1024
#define BS     1024
#define BN_EPS 1e-5f

typedef __attribute__((ext_vector_type(16))) _Float16 v16h;
typedef __attribute__((ext_vector_type(8)))  _Float16 v8h;
typedef __attribute__((ext_vector_type(8)))  float    v8f;
typedef int v4i_ __attribute__((vector_size(16)));

#define CAT16(lo, hi) __builtin_shufflevector(lo, hi, 0, 1, 2, 3, 4, 5, 6, 7, \
                                              8, 9, 10, 11, 12, 13, 14, 15)
#define WMMA_F16(a, b, c) __builtin_amdgcn_wmma_f32_16x16x32_f16(false, a, false, b, \
                                                                 (short)0, c, false, false)

// Load a 16-element f16 fragment: two b128 loads (lo at p, hi at p+16 elems)
__device__ __forceinline__ v16h load_frag(const _Float16* p) {
  v8h lo = *reinterpret_cast<const v8h*>(p);
  v8h hi = *reinterpret_cast<const v8h*>(p + 16);
  return CAT16(lo, hi);
}

// ----- optional CDNA5 async global->LDS path (guarded; fallback compiles) ---
#if defined(__has_builtin)
#if __has_builtin(__builtin_amdgcn_global_load_async_to_lds_b128)
#define HAVE_ASYNC_LDS 1
#endif
#endif

#ifdef HAVE_ASYNC_LDS
__device__ __forceinline__ void async_ld16(void* lds, const void* g) {
  __builtin_amdgcn_global_load_async_to_lds_b128(
      (__attribute__((address_space(1))) v4i_*)(unsigned long long)g,
      (__attribute__((address_space(3))) v4i_*)(unsigned)(unsigned long long)lds,
      0, 0);
}
#define ASYNC_WAIT_1() asm volatile("s_wait_asynccnt 0x1" ::: "memory")
#define ASYNC_WAIT_0() asm volatile("s_wait_asynccnt 0x0" ::: "memory")
#endif

// ---------------------------------------------------------------------------
// Pack: f32 -> f16, 8 elements per thread (b128 in, b128 out)
// ---------------------------------------------------------------------------
__global__ void __launch_bounds__(256)
pack_f16_kernel(const float* __restrict__ src, _Float16* __restrict__ dst, int n) {
  int i = (blockIdx.x * blockDim.x + threadIdx.x) * 8;
  if (i >= n) return;
  float4 a = *reinterpret_cast<const float4*>(src + i);
  float4 b = *reinterpret_cast<const float4*>(src + i + 4);
  v8h o;
  o[0] = (_Float16)a.x; o[1] = (_Float16)a.y; o[2] = (_Float16)a.z; o[3] = (_Float16)a.w;
  o[4] = (_Float16)b.x; o[5] = (_Float16)b.y; o[6] = (_Float16)b.z; o[7] = (_Float16)b.w;
  *reinterpret_cast<v8h*>(dst + i) = o;
}

// ---------------------------------------------------------------------------
// Pack normalized Q: qn = Q*scale + shift -> f16
// ---------------------------------------------------------------------------
__global__ void __launch_bounds__(256)
pack_qn_kernel(const float* __restrict__ Q, const float* __restrict__ scale,
               const float* __restrict__ shift, _Float16* __restrict__ qn) {
  int i = (blockIdx.x * blockDim.x + threadIdx.x) * 8;
  if (i >= BS * F_DIM) return;
  int f = i & (F_DIM - 1);
  float4 a  = *reinterpret_cast<const float4*>(Q + i);
  float4 b  = *reinterpret_cast<const float4*>(Q + i + 4);
  float4 s0 = *reinterpret_cast<const float4*>(scale + f);
  float4 s1 = *reinterpret_cast<const float4*>(scale + f + 4);
  float4 h0 = *reinterpret_cast<const float4*>(shift + f);
  float4 h1 = *reinterpret_cast<const float4*>(shift + f + 4);
  v8h o;
  o[0] = (_Float16)(a.x * s0.x + h0.x); o[1] = (_Float16)(a.y * s0.y + h0.y);
  o[2] = (_Float16)(a.z * s0.z + h0.z); o[3] = (_Float16)(a.w * s0.w + h0.w);
  o[4] = (_Float16)(b.x * s1.x + h1.x); o[5] = (_Float16)(b.y * s1.y + h1.y);
  o[6] = (_Float16)(b.z * s1.z + h1.z); o[7] = (_Float16)(b.w * s1.w + h1.w);
  *reinterpret_cast<v8h*>(qn + i) = o;
}

// ---------------------------------------------------------------------------
// Kernel 1: Q = X @ Wq^T + bq  (f16 WMMA, f32 acc). 16x64 tile per wave.
// Register double-buffered: loads of step k+1 overlap WMMAs of step k.
// ---------------------------------------------------------------------------
__global__ void __launch_bounds__(32)
gemm_q_kernel(const _Float16* __restrict__ XH, const _Float16* __restrict__ WH,
              const float* __restrict__ bq, float* __restrict__ Q) {
  const int lane  = threadIdx.x & 31;
  const int mBase = blockIdx.x * 16;
  const int nBase = blockIdx.y * 64;
  const int lrow  = lane & 15;
  const int lhi   = (lane >> 4) * 8;   // K sub-group per ISA A/B layout

  const _Float16* xrow = XH + (size_t)(mBase + lrow) * IN_DIM + lhi;
  const _Float16* wrow = WH + (size_t)(nBase + lrow) * IN_DIM + lhi;  // B[k][n]=Wq[n][k]

  v8f acc[4] = {v8f{}, v8f{}, v8f{}, v8f{}};
  v16h a  = load_frag(xrow);
  v16h b0 = load_frag(wrow);
  v16h b1 = load_frag(wrow + 16 * IN_DIM);
  v16h b2 = load_frag(wrow + 32 * IN_DIM);
  v16h b3 = load_frag(wrow + 48 * IN_DIM);
  for (int k = 32; k < IN_DIM; k += 32) {
    v16h an = load_frag(xrow + k);
    v16h c0 = load_frag(wrow + k);
    v16h c1 = load_frag(wrow + 16 * IN_DIM + k);
    v16h c2 = load_frag(wrow + 32 * IN_DIM + k);
    v16h c3 = load_frag(wrow + 48 * IN_DIM + k);
    acc[0] = WMMA_F16(a, b0, acc[0]);
    acc[1] = WMMA_F16(a, b1, acc[1]);
    acc[2] = WMMA_F16(a, b2, acc[2]);
    acc[3] = WMMA_F16(a, b3, acc[3]);
    a = an; b0 = c0; b1 = c1; b2 = c2; b3 = c3;
  }
  acc[0] = WMMA_F16(a, b0, acc[0]);
  acc[1] = WMMA_F16(a, b1, acc[1]);
  acc[2] = WMMA_F16(a, b2, acc[2]);
  acc[3] = WMMA_F16(a, b3, acc[3]);

  const int mOff = (lane >> 4) * 8;  // C/D layout: lanes>=16 hold M+8
#pragma unroll
  for (int j = 0; j < 4; ++j) {
    const int   n    = nBase + j * 16 + lrow;
    const float bias = bq[n];
#pragma unroll
    for (int r = 0; r < 8; ++r)
      Q[(size_t)(mBase + mOff + r) * F_DIM + n] = acc[j][r] + bias;
  }
}

// ---------------------------------------------------------------------------
// Kernel 2: batchnorm stats -> per-feature scale/shift
// ---------------------------------------------------------------------------
__global__ void bn_stats_kernel(const float* __restrict__ Q,
                                const float* __restrict__ gamma,
                                const float* __restrict__ beta,
                                float* __restrict__ scale,
                                float* __restrict__ shift) {
  int f = blockIdx.x * blockDim.x + threadIdx.x;
  if (f >= F_DIM) return;
  float s = 0.f, ss = 0.f;
  for (int b = 0; b < BS; ++b) {
    float v = Q[(size_t)b * F_DIM + f];
    s += v; ss += v * v;
  }
  float mu  = s / (float)BS;
  float var = ss / (float)BS - mu * mu;
  float sc  = rsqrtf(var + BN_EPS) * gamma[f];
  scale[f] = sc;
  shift[f] = beta[f] - mu * sc;
}

// ---------------------------------------------------------------------------
// Kernel 3: S[half][b][h][key] = qn . keys  (16x64 per wave, K=128 unrolled,
// register double-buffered)
// ---------------------------------------------------------------------------
__global__ void __launch_bounds__(32)
score_kernel(const _Float16* __restrict__ qnH, const _Float16* __restrict__ keysH,
             float* __restrict__ S) {
  const int lane  = threadIdx.x & 31;
  const int mBase = blockIdx.x * 16;           // batch rows
  const int nBase = blockIdx.y * 64;           // keys
  const int hh    = blockIdx.z;                // h*2 + half
  const int h     = hh >> 1, half = hh & 1;
  const int f0    = h * KDIM + half * HALF;

  const int lrow = lane & 15;
  const int lhi  = (lane >> 4) * 8;

  const _Float16* qrow = qnH + (size_t)(mBase + lrow) * F_DIM + f0 + lhi;
  const _Float16* krow = keysH + ((size_t)(h * 2 + half) * NKEYS + (nBase + lrow)) * HALF + lhi;

  v8f acc[4] = {v8f{}, v8f{}, v8f{}, v8f{}};
  v16h a  = load_frag(qrow);
  v16h b0 = load_frag(krow);
  v16h b1 = load_frag(krow + 16 * HALF);
  v16h b2 = load_frag(krow + 32 * HALF);
  v16h b3 = load_frag(krow + 48 * HALF);
#pragma unroll
  for (int k = 32; k < HALF; k += 32) {
    v16h an = load_frag(qrow + k);
    v16h c0 = load_frag(krow + k);
    v16h c1 = load_frag(krow + 16 * HALF + k);
    v16h c2 = load_frag(krow + 32 * HALF + k);
    v16h c3 = load_frag(krow + 48 * HALF + k);
    acc[0] = WMMA_F16(a, b0, acc[0]);
    acc[1] = WMMA_F16(a, b1, acc[1]);
    acc[2] = WMMA_F16(a, b2, acc[2]);
    acc[3] = WMMA_F16(a, b3, acc[3]);
    a = an; b0 = c0; b1 = c1; b2 = c2; b3 = c3;
  }
  acc[0] = WMMA_F16(a, b0, acc[0]);
  acc[1] = WMMA_F16(a, b1, acc[1]);
  acc[2] = WMMA_F16(a, b2, acc[2]);
  acc[3] = WMMA_F16(a, b3, acc[3]);

  const int mOff = (lane >> 4) * 8;
#pragma unroll
  for (int j = 0; j < 4; ++j) {
    const int n = nBase + j * 16 + lrow;
#pragma unroll
    for (int r = 0; r < 8; ++r) {
      int b = mBase + mOff + r;
      S[(((size_t)half * BS + b) * HEADS + h) * NKEYS + n] = acc[j][r];
    }
  }
}

// ---------------------------------------------------------------------------
// Kernel 4: top-32 of 512, one thread per (half,b,h) task (8192 tasks).
// ---------------------------------------------------------------------------
__global__ void topk1_kernel(const float* __restrict__ S,
                             float* __restrict__ tv, int* __restrict__ ti) {
  int t = blockIdx.x * blockDim.x + threadIdx.x;
  if (t >= 2 * BS * HEADS) return;
  const float* s = S + (size_t)t * NKEYS;
  float val[KNN]; int ind[KNN];
#pragma unroll
  for (int i = 0; i < KNN; ++i) { val[i] = -3.4e38f; ind[i] = 0; }
  for (int k = 0; k < NKEYS; ++k) {
    float v = s[k];
    if (v > val[KNN - 1]) {
      int p = KNN - 1;
      while (p > 0 && val[p - 1] < v) { val[p] = val[p - 1]; ind[p] = ind[p - 1]; --p; }
      val[p] = v; ind[p] = k;
    }
  }
  float* ov = tv + (size_t)t * KNN;
  int*   oi = ti + (size_t)t * KNN;
  for (int i = 0; i < KNN; ++i) { ov[i] = val[i]; oi[i] = ind[i]; }
}

// ---------------------------------------------------------------------------
// Kernel 5: cartesian 32x32 sums -> top-32 -> softmax. One thread per (b,h).
// ---------------------------------------------------------------------------
__global__ void topk2_kernel(const float* __restrict__ tv, const int* __restrict__ ti,
                             float* __restrict__ w, int* __restrict__ idx) {
  int t = blockIdx.x * blockDim.x + threadIdx.x;   // t = b*HEADS + h
  if (t >= BS * HEADS) return;
  const float* s1 = tv + (size_t)t * KNN;                       // half 0
  const int*   i1 = ti + (size_t)t * KNN;
  const float* s2 = tv + ((size_t)BS * HEADS + t) * KNN;        // half 1
  const int*   i2 = ti + ((size_t)BS * HEADS + t) * KNN;

  float val[KNN]; int ind[KNN];
#pragma unroll
  for (int i = 0; i < KNN; ++i) { val[i] = -3.4e38f; ind[i] = 0; }
  for (int i = 0; i < KNN; ++i) {
    float a  = s1[i];
    int   ia = i1[i] * NKEYS;
    for (int j = 0; j < KNN; ++j) {
      float v = a + s2[j];
      if (v > val[KNN - 1]) {
        int p = KNN - 1;
        while (p > 0 && val[p - 1] < v) { val[p] = val[p - 1]; ind[p] = ind[p - 1]; --p; }
        val[p] = v; ind[p] = ia + i2[j];
      }
    }
  }
  float m = val[0], sum = 0.f, e[KNN];
  for (int i = 0; i < KNN; ++i) { e[i] = expf(val[i] - m); sum += e[i]; }
  float inv = 1.f / sum;
  float* ow = w   + (size_t)t * KNN;
  int*   oi = idx + (size_t)t * KNN;
  for (int i = 0; i < KNN; ++i) { ow[i] = e[i] * inv; oi[i] = ind[i]; }
}

// ---------------------------------------------------------------------------
// Kernel 6: out[b,:] = sum_n w[b,n] * values[idx[b,n], :]
// HBM-bound gather (~512 MB @ 23.3 TB/s). Async path: triple-buffered LDS
// stream via GLOBAL_LOAD_ASYNC_TO_LDS_B128 + s_wait_asynccnt (per-lane 16B
// chunks; in-order async completion => wait<=1 guarantees row n landed).
// ---------------------------------------------------------------------------
#ifdef HAVE_ASYNC_LDS
__global__ void __launch_bounds__(256)
gather_kernel(const float* __restrict__ w, const int* __restrict__ idx,
              const float* __restrict__ values, float* __restrict__ out) {
  __shared__ float sw[HEADS * KNN];
  __shared__ int   si[HEADS * KNN];
  __shared__ float buf[3][VDIM];          // 12 KB triple buffer
  const int b = blockIdx.x;
  const int t = threadIdx.x;
  if (t < HEADS * KNN) {
    sw[t] = w[(size_t)b * HEADS * KNN + t];
    si[t] = idx[(size_t)b * HEADS * KNN + t];
  }
  __syncthreads();
  const int v0 = t * 4;
  async_ld16(&buf[0][v0], values + (size_t)si[0] * VDIM + v0);   // prime row 0
  float4 acc = make_float4(0.f, 0.f, 0.f, 0.f);
  for (int n = 0; n < HEADS * KNN; ++n) {
    if (n + 1 < HEADS * KNN) {
      async_ld16(&buf[(n + 1) % 3][v0], values + (size_t)si[n + 1] * VDIM + v0);
      ASYNC_WAIT_1();   // row n complete (row n+1 may stay in flight)
    } else {
      ASYNC_WAIT_0();
    }
    const float4 vv = *reinterpret_cast<const float4*>(&buf[n % 3][v0]);
    const float wn = sw[n];
    acc.x += wn * vv.x; acc.y += wn * vv.y;
    acc.z += wn * vv.z; acc.w += wn * vv.w;
  }
  *reinterpret_cast<float4*>(out + (size_t)b * VDIM + v0) = acc;
}
#else
__global__ void __launch_bounds__(256)
gather_kernel(const float* __restrict__ w, const int* __restrict__ idx,
              const float* __restrict__ values, float* __restrict__ out) {
  __shared__ float sw[HEADS * KNN];
  __shared__ int   si[HEADS * KNN];
  const int b = blockIdx.x;
  const int t = threadIdx.x;
  if (t < HEADS * KNN) {
    sw[t] = w[(size_t)b * HEADS * KNN + t];
    si[t] = idx[(size_t)b * HEADS * KNN + t];
  }
  __syncthreads();
  const int v0 = t * 4;
  float4 acc = make_float4(0.f, 0.f, 0.f, 0.f);
  for (int n = 0; n < HEADS * KNN; ++n) {
    if (n + 4 < HEADS * KNN)
      __builtin_prefetch(values + (size_t)si[n + 4] * VDIM + v0, 0, 0);
    const float4 vv = *reinterpret_cast<const float4*>(values + (size_t)si[n] * VDIM + v0);
    const float wn = sw[n];
    acc.x += wn * vv.x; acc.y += wn * vv.y;
    acc.z += wn * vv.z; acc.w += wn * vv.w;
  }
  *reinterpret_cast<float4*>(out + (size_t)b * VDIM + v0) = acc;
}
#endif

// ---------------------------------------------------------------------------
extern "C" void kernel_launch(void* const* d_in, const int* in_sizes, int n_in,
                              void* d_out, int out_size, void* d_ws, size_t ws_size,
                              hipStream_t stream) {
  (void)in_sizes; (void)n_in; (void)out_size; (void)ws_size;
  const float* x      = (const float*)d_in[0];
  const float* Wq     = (const float*)d_in[1];
  const float* bq     = (const float*)d_in[2];
  const float* gamma  = (const float*)d_in[3];
  const float* beta   = (const float*)d_in[4];
  const float* keys   = (const float*)d_in[5];
  const float* values = (const float*)d_in[6];
  float* out = (float*)d_out;

  char*  ws  = (char*)d_ws;
  size_t off = 0;
  auto alloc = [&](size_t bytes) -> void* {
    void* p = ws + off;
    off += (bytes + 255) & ~(size_t)255;
    return p;
  };

  const int NKEY_ELEMS = HEADS * 2 * NKEYS * HALF;  // 524288

  float*    Q     = (float*)   alloc((size_t)BS * F_DIM * sizeof(float));            // 4 MB
  float*    scale = (float*)   alloc((size_t)F_DIM * sizeof(float));
  float*    shift = (float*)   alloc((size_t)F_DIM * sizeof(float));
  float*    S     = (float*)   alloc((size_t)2 * BS * HEADS * NKEYS * sizeof(float)); // 16 MB
  float*    tv    = (float*)   alloc((size_t)2 * BS * HEADS * KNN * sizeof(float));
  int*      ti    = (int*)     alloc((size_t)2 * BS * HEADS * KNN * sizeof(int));
  float*    w     = (float*)   alloc((size_t)BS * HEADS * KNN * sizeof(float));
  int*      idx   = (int*)     alloc((size_t)BS * HEADS * KNN * sizeof(int));
  _Float16* XH    = (_Float16*)alloc((size_t)BS * IN_DIM * sizeof(_Float16));        // 2 MB
  _Float16* WH    = (_Float16*)alloc((size_t)F_DIM * IN_DIM * sizeof(_Float16));     // 2 MB
  _Float16* qnH   = (_Float16*)alloc((size_t)BS * F_DIM * sizeof(_Float16));         // 2 MB
  _Float16* keysH = (_Float16*)alloc((size_t)NKEY_ELEMS * sizeof(_Float16));         // 1 MB

  // 0) pack inputs to f16 (one-time conversion, bandwidth-bound)
  pack_f16_kernel<<<dim3((BS * IN_DIM / 8) / 256), 256, 0, stream>>>(x, XH, BS * IN_DIM);
  pack_f16_kernel<<<dim3((F_DIM * IN_DIM / 8) / 256), 256, 0, stream>>>(Wq, WH, F_DIM * IN_DIM);
  pack_f16_kernel<<<dim3((NKEY_ELEMS / 8) / 256), 256, 0, stream>>>(keys, keysH, NKEY_ELEMS);
  // 1) Q = X @ Wq^T + bq  (WMMA, 16x64 tiles per wave, double-buffered)
  gemm_q_kernel<<<dim3(BS / 16, F_DIM / 64), 32, 0, stream>>>(XH, WH, bq, Q);
  // 2) batchnorm stats
  bn_stats_kernel<<<dim3(F_DIM / 256), 256, 0, stream>>>(Q, gamma, beta, scale, shift);
  // 2b) normalized Q -> f16
  pack_qn_kernel<<<dim3((BS * F_DIM / 8) / 256), 256, 0, stream>>>(Q, scale, shift, qnH);
  // 3) sub-key scores (WMMA, 16x64 tiles per wave, double-buffered)
  score_kernel<<<dim3(BS / 16, NKEYS / 64, HEADS * 2), 32, 0, stream>>>(qnH, keysH, S);
  // 4) per-half top-32
  topk1_kernel<<<dim3((2 * BS * HEADS) / 256), 256, 0, stream>>>(S, tv, ti);
  // 5) combined top-32 + softmax
  topk2_kernel<<<dim3((BS * HEADS) / 256), 256, 0, stream>>>(tv, ti, w, idx);
  // 6) gather-weighted sum into output
  gather_kernel<<<dim3(BS), 256, 0, stream>>>(w, idx, values, out);
}
